// AttnBlock_45019847197419
// MI455X (gfx1250) — matrix operand order repaired and checked by reading, not compiled, not run on hardware
//
#include <hip/hip_runtime.h>

// CDNA5 / gfx1250 flash-attention block. wave32, WMMA f16 -> f32.
// v2: V transposed in LDS (all B-frags contiguous -> ds_load_b128),
//     double-buffered K/V tiles (1 barrier/iter), per-wave P fence only.
typedef __attribute__((ext_vector_type(16))) _Float16 v16h;
typedef __attribute__((ext_vector_type(8)))  float    v8f;

#define BATCH 4
#define CH    64
#define NTOK  4096   // 64*64 tokens
#define BQ    128    // query rows per workgroup (8 waves x 16)
#define BK    64     // key tile per iteration
#define LDSTR 72     // padded half-stride for 64-wide LDS tiles
#define PSTR  68     // padded half-stride for 16x64 P tile

// ---- WMMA fragment builders (wave32; contiguous k = 16h + s, same in A and B) ----

// A-matrix 16x32 f16: lane m = lane&15; k-base 0 (lanes 0-15) / 16 (lanes 16-31);
// halves 0-15 -> k=kb+0..15 (contiguous).
static __device__ __forceinline__ v16h ld_afrag(const _Float16* p0, int stride,
                                                int m0, int k0) {
  int lane = threadIdx.x & 31;
  int m  = m0 + (lane & 15);
  int kb = (lane < 16) ? 0 : 16;
  const _Float16* p = p0 + m * stride + k0 + kb;
  v16h a;
#pragma unroll
  for (int i = 0; i < 8; ++i) a[i] = p[i];
#pragma unroll
  for (int i = 0; i < 8; ++i) a[8 + i] = p[8 + i];
  return a;
}

// B-matrix 32x16 f16 from storage laid out [n][k] (B transposed in memory):
// lane n = lane&15; k-base 0 / 16 by lane half; halves cover k contiguous.
static __device__ __forceinline__ v16h ld_bfrag_nk(const _Float16* p0, int stride,
                                                   int n0, int k0) {
  int lane = threadIdx.x & 31;
  int n  = n0 + (lane & 15);
  int kb = (lane < 16) ? 0 : 16;
  const _Float16* p = p0 + n * stride + k0 + kb;
  v16h b;
#pragma unroll
  for (int i = 0; i < 16; ++i) b[i] = p[i];
  return b;
}

static __device__ __forceinline__ v8f wmma_f16(v16h a, v16h b, v8f c) {
  return __builtin_amdgcn_wmma_f32_16x16x32_f16(false, a, false, b,
                                                (short)0, c, false, false);
}

// ---------------- Kernel 1: Q/K/V projections (1x1 convs) ----------------
__global__ __launch_bounds__(256) void qkv_proj_kernel(
    const float* __restrict__ x, const float* __restrict__ Wq,
    const float* __restrict__ Wk, const float* __restrict__ Wv,
    _Float16* __restrict__ Qg, _Float16* __restrict__ Kg,
    _Float16* __restrict__ Vg) {
  __shared__ _Float16 xt[BQ * LDSTR];          // x tile transposed: [token][chan]
  __shared__ _Float16 wsh[3][CH * LDSTR];      // weights [o][c]
  const int b  = blockIdx.y;
  const int n0 = blockIdx.x * BQ;
  const int tid = threadIdx.x;

  const float* xb = x + (size_t)b * CH * NTOK;
  for (int e = tid; e < CH * BQ; e += 256) {   // coalesced global, scatter to LDS
    int c = e >> 7;            // /BQ
    int t = e & (BQ - 1);
    xt[t * LDSTR + c] = (_Float16)xb[c * NTOK + n0 + t];
  }
  const float* wsrc[3] = {Wq, Wk, Wv};
#pragma unroll
  for (int m = 0; m < 3; ++m)
    for (int e = tid; e < CH * CH; e += 256) {
      int o = e >> 6, c = e & 63;
      wsh[m][o * LDSTR + c] = (_Float16)wsrc[m][e];
    }
  __syncthreads();

  const int w = tid >> 5, lane = tid & 31;
  const int col = lane & 15, rbase = (lane >> 4) * 8;
  const int t0 = w * 16;
  v16h a0 = ld_afrag(xt, LDSTR, t0, 0);
  v16h a1 = ld_afrag(xt, LDSTR, t0, 32);

  _Float16* dst[3] = {Qg, Kg, Vg};
#pragma unroll
  for (int m = 0; m < 3; ++m) {
#pragma unroll
    for (int nt = 0; nt < 4; ++nt) {
      v8f acc = {};
      acc = wmma_f16(a0, ld_bfrag_nk(wsh[m], LDSTR, nt * 16, 0),  acc);
      acc = wmma_f16(a1, ld_bfrag_nk(wsh[m], LDSTR, nt * 16, 32), acc);
      _Float16* d = dst[m] + ((size_t)b * NTOK + n0 + t0) * CH + nt * 16 + col;
#pragma unroll
      for (int i = 0; i < 8; ++i)
        d[(size_t)(rbase + i) * CH] = (_Float16)acc[i];
    }
  }
}

// cooperative K/V tile loader: K row-major [key][chan], V transposed [chan][key]
static __device__ __forceinline__ void load_kv_tile(
    const _Float16* __restrict__ Kg, const _Float16* __restrict__ Vg,
    size_t base, _Float16* kbuf, _Float16* vbufT, int tid) {
  const uint2* ksrc = (const uint2*)(Kg + base);
  const uint2* vsrc = (const uint2*)(Vg + base);
  for (int chk = tid; chk < BK * (CH / 4); chk += 256) {
    int r = chk >> 4, cc = chk & 15;           // key row, 4-chan chunk
    *(uint2*)&kbuf[r * LDSTR + cc * 4] = ksrc[chk];
    uint2 vv = vsrc[chk];
    const _Float16* vh = (const _Float16*)&vv;
#pragma unroll
    for (int j = 0; j < 4; ++j)
      vbufT[(cc * 4 + j) * LDSTR + r] = vh[j]; // transpose into [chan][key]
  }
}

// ---------------- Kernel 2: flash attention (online softmax) ----------------
__global__ __launch_bounds__(256) void flash_attn_kernel(
    const _Float16* __restrict__ Qg, const _Float16* __restrict__ Kg,
    const _Float16* __restrict__ Vg, _Float16* __restrict__ Hg) {
  __shared__ _Float16 ksh[2][BK * LDSTR];      // [key][chan], double-buffered
  __shared__ _Float16 vshT[2][CH * LDSTR];     // [chan][key], double-buffered
  __shared__ _Float16 psh[8][16 * PSTR];       // per-wave P staging (C->A layout)
  const int b  = blockIdx.y;
  const int n0 = blockIdx.x * BQ;
  const int tid = threadIdx.x;
  const int w = tid >> 5, lane = tid & 31;
  const int col = lane & 15, rbase = (lane >> 4) * 8;
  const int qrow0 = n0 + w * 16;

  const _Float16* qp = Qg + ((size_t)b * NTOK + qrow0) * CH;
  v16h qa0 = ld_afrag(qp, CH, 0, 0);
  v16h qa1 = ld_afrag(qp, CH, 0, 32);

  v8f oacc[4];
#pragma unroll
  for (int t = 0; t < 4; ++t) oacc[t] = (v8f){};
  float mrow[8], lrow[8];
#pragma unroll
  for (int i = 0; i < 8; ++i) { mrow[i] = -3.0e38f; lrow[i] = 0.f; }
  const float scale = 0.125f;                  // C^-0.5, C=64

  const size_t kvbase = (size_t)b * NTOK * CH;
  load_kv_tile(Kg, Vg, kvbase, ksh[0], vshT[0], tid);   // prologue: tile 0
  __syncthreads();

  const int NT = NTOK / BK;
  for (int it = 0; it < NT; ++it) {
    const int cur = it & 1;
    if (it + 1 < NT)   // prefetch next tile into alternate buffer (overlaps WMMA)
      load_kv_tile(Kg, Vg, kvbase + (size_t)(it + 1) * BK * CH,
                   ksh[cur ^ 1], vshT[cur ^ 1], tid);
    const _Float16* kb = ksh[cur];
    const _Float16* vb = vshT[cur];

    // S = Q * K^T  (4 tiles of 16 keys)
    v8f s[4];
#pragma unroll
    for (int js = 0; js < 4; ++js) {
      v8f t = {};
      t = wmma_f16(qa0, ld_bfrag_nk(kb, LDSTR, js * 16, 0),  t);
      t = wmma_f16(qa1, ld_bfrag_nk(kb, LDSTR, js * 16, 32), t);
      s[js] = t * scale;
    }

    // online softmax: rows live at fixed VGPR slot across a 16-lane half
    float alpha[8];
#pragma unroll
    for (int i = 0; i < 8; ++i) {
      float mx = fmaxf(fmaxf(s[0][i], s[1][i]), fmaxf(s[2][i], s[3][i]));
      mx = fmaxf(mx, mrow[i]);
#pragma unroll
      for (int d = 1; d < 16; d <<= 1) mx = fmaxf(mx, __shfl_xor(mx, d, 32));
      alpha[i] = __expf(mrow[i] - mx);
      mrow[i] = mx;
      float sum = 0.f;
#pragma unroll
      for (int js = 0; js < 4; ++js) {
        float p = __expf(s[js][i] - mx);
        s[js][i] = p;
        sum += p;
      }
#pragma unroll
      for (int d = 1; d < 16; d <<= 1) sum += __shfl_xor(sum, d, 32);
      lrow[i] = lrow[i] * alpha[i] + sum;
#pragma unroll
      for (int t = 0; t < 4; ++t) oacc[t][i] *= alpha[i];
    }

    // P: C-layout regs -> per-wave LDS region -> A-layout fragments.
    // Same-wave LDS ops are in-order (DScnt); only a compiler fence is needed.
    _Float16* pw = psh[w];
#pragma unroll
    for (int i = 0; i < 8; ++i) {
      int r = rbase + i;
#pragma unroll
      for (int js = 0; js < 4; ++js)
        pw[r * PSTR + js * 16 + col] = (_Float16)s[js][i];
    }
    asm volatile("s_wait_dscnt 0" ::: "memory");
    v16h pa0 = ld_afrag(pw, PSTR, 0, 0);
    v16h pa1 = ld_afrag(pw, PSTR, 0, 32);

    // O += P * V   (V read as contiguous [chan][key] B-frags)
#pragma unroll
    for (int t = 0; t < 4; ++t) {
      oacc[t] = wmma_f16(pa0, ld_bfrag_nk(vb, LDSTR, t * 16, 0),  oacc[t]);
      oacc[t] = wmma_f16(pa1, ld_bfrag_nk(vb, LDSTR, t * 16, 32), oacc[t]);
    }
    __syncthreads();   // single barrier: publishes next tile, retires this one
  }

  // normalize and store H (f16 token-major)
#pragma unroll
  for (int i = 0; i < 8; ++i) {
    float inv = 1.0f / lrow[i];
#pragma unroll
    for (int t = 0; t < 4; ++t) oacc[t][i] *= inv;
  }
  _Float16* hp = Hg + ((size_t)b * NTOK + qrow0) * CH + col;
#pragma unroll
  for (int i = 0; i < 8; ++i) {
    int r = rbase + i;
#pragma unroll
    for (int t = 0; t < 4; ++t)
      hp[(size_t)r * CH + t * 16] = (_Float16)oacc[t][i];
  }
}

// ---------------- Kernel 3: output projection + residual ----------------
__global__ __launch_bounds__(256) void out_proj_kernel(
    const float* __restrict__ x, const float* __restrict__ Wp,
    const _Float16* __restrict__ Hg, float* __restrict__ out) {
  __shared__ _Float16 hsh[BQ * LDSTR];
  __shared__ _Float16 wsh[CH * LDSTR];
  const int b  = blockIdx.y;
  const int n0 = blockIdx.x * BQ;
  const int tid = threadIdx.x;

  const uint2* hsrc = (const uint2*)(Hg + ((size_t)b * NTOK + n0) * CH);
  for (int chk = tid; chk < BQ * (CH / 4); chk += 256) {
    int r = chk >> 4, cc = chk & 15;
    *(uint2*)&hsh[r * LDSTR + cc * 4] = hsrc[chk];
  }
  for (int e = tid; e < CH * CH; e += 256) {
    int o = e >> 6, c = e & 63;
    wsh[o * LDSTR + c] = (_Float16)Wp[e];
  }
  __syncthreads();

  const int w = tid >> 5, lane = tid & 31;
  const int col = lane & 15, rbase = (lane >> 4) * 8;
  const int t0 = w * 16;
  v16h a0 = ld_afrag(hsh, LDSTR, t0, 0);
  v16h a1 = ld_afrag(hsh, LDSTR, t0, 32);

#pragma unroll
  for (int nt = 0; nt < 4; ++nt) {
    v8f acc = {};
    acc = wmma_f16(a0, ld_bfrag_nk(wsh, LDSTR, nt * 16, 0),  acc);
    acc = wmma_f16(a1, ld_bfrag_nk(wsh, LDSTR, nt * 16, 32), acc);
    int co = nt * 16 + col;
    const float* xp = x   + ((size_t)b * CH + co) * NTOK + n0 + t0;
    float*       op = out + ((size_t)b * CH + co) * NTOK + n0 + t0;
#pragma unroll
    for (int i = 0; i < 8; ++i) {
      int r = rbase + i;
      op[r] = xp[r] + acc[i];   // residual
    }
  }
}

extern "C" void kernel_launch(void* const* d_in, const int* in_sizes, int n_in,
                              void* d_out, int out_size, void* d_ws, size_t ws_size,
                              hipStream_t stream) {
  const float* x  = (const float*)d_in[0];
  const float* Wq = (const float*)d_in[1];
  const float* Wk = (const float*)d_in[2];
  const float* Wv = (const float*)d_in[3];
  const float* Wp = (const float*)d_in[4];

  // workspace: Q,K,V,H f16 [B][N][C] = 2 MB each (8 MB total)
  const size_t tsz = (size_t)BATCH * NTOK * CH;
  _Float16* Qg = (_Float16*)d_ws;
  _Float16* Kg = Qg + tsz;
  _Float16* Vg = Kg + tsz;
  _Float16* Hg = Vg + tsz;

  dim3 grid(NTOK / BQ, BATCH), blk(256);
  qkv_proj_kernel  <<<grid, blk, 0, stream>>>(x, Wq, Wk, Wv, Qg, Kg, Vg);
  flash_attn_kernel<<<grid, blk, 0, stream>>>(Qg, Kg, Vg, Hg);
  out_proj_kernel  <<<grid, blk, 0, stream>>>(x, Wp, Hg, (float*)d_out);
}
